// BidirectionalPropagation_22591527977132
// MI455X (gfx1250) — matrix operand order repaired and checked by reading, not compiled, not run on hardware
//
#include <hip/hip_runtime.h>
#include <cstdint>
#include <cstddef>

// ---------------------------------------------------------------------------
// Problem constants (match reference): B=1, T=8, C=64, H=W=96, DG=16
// ---------------------------------------------------------------------------
#define TV   8
#define CV   64
#define HV   96
#define WV   96
#define HWC  (HV*WV)          // 9216 (multiple of 64)
#define CHWn (CV*HWC)         // 589824 halves per C-channel feature map
#define DGV  16

typedef __attribute__((ext_vector_type(16))) _Float16 v16h;
typedef __attribute__((ext_vector_type(8)))  float    v8f;
typedef __attribute__((ext_vector_type(4)))  int      v4i;

// gfx1250 async global->LDS path (ASYNCcnt), guarded so absence still compiles
#if defined(__HIP_DEVICE_COMPILE__)
#if __has_builtin(__builtin_amdgcn_global_load_async_to_lds_b128) && \
    __has_builtin(__builtin_amdgcn_s_wait_asynccnt)
#define ASYNC_LDS 1
typedef __attribute__((address_space(1))) v4i gv4i;   // global int4
typedef __attribute__((address_space(3))) v4i lv4i;   // LDS int4
#endif
#endif

// ---------------------------------------------------------------------------
// Elementwise helpers
// ---------------------------------------------------------------------------
__global__ __launch_bounds__(256) void k_cvt16(const float* __restrict__ s,
                                               _Float16* __restrict__ d, int n) {
  int i = blockIdx.x * 256 + threadIdx.x;
  if (i < n) d[i] = (_Float16)s[i];
}

// convert weights f32 -> f16 with zero padding of extra rows (row-major [rows][K])
__global__ __launch_bounds__(256) void k_cvtw(const float* __restrict__ s,
                                              _Float16* __restrict__ d,
                                              int nvalid, int ntot) {
  int i = blockIdx.x * 256 + threadIdx.x;
  if (i < ntot) d[i] = (i < nvalid) ? (_Float16)s[i] : (_Float16)0.f;
}

__global__ __launch_bounds__(256) void k_copy8(_Float16* __restrict__ d,
                                               const _Float16* __restrict__ s, int n8) {
  int i = blockIdx.x * 256 + threadIdx.x;
  if (i < n8) ((uint4*)d)[i] = ((const uint4*)s)[i];
}

__global__ __launch_bounds__(256) void k_fill8(_Float16* __restrict__ d, int n8) {
  int i = blockIdx.x * 256 + threadIdx.x;
  uint4 z; z.x = 0; z.y = 0; z.z = 0; z.w = 0;
  if (i < n8) ((uint4*)d)[i] = z;
}

// out4 (432 ch, f16) -> offsets = 5*tanh (288 ch f32), mask = sigmoid (144 ch f32)
__global__ __launch_bounds__(256) void k_offmask(const _Float16* __restrict__ o4,
                                                 float* __restrict__ offs,
                                                 float* __restrict__ msk) {
  int i = blockIdx.x * 256 + threadIdx.x;
  if (i >= 432 * HWC) return;
  float v = (float)o4[i];
  int ch = i / HWC;
  int p  = i - ch * HWC;
  if (ch < 288) offs[i] = 5.0f * tanhf(v);
  else          msk[(ch - 288) * HWC + p] = 1.0f / (1.0f + expf(-v));
}

// Build modulated-deformable im2col columns: cols[(g*8+c)*9 + k][p] (f16).
// Sample position shared across the 8 channels of each deform group.
__global__ __launch_bounds__(256) void k_cols(const _Float16* __restrict__ xin,
                                              const float* __restrict__ offs,
                                              const float* __restrict__ msk,
                                              _Float16* __restrict__ cols) {
  int i = blockIdx.x * 256 + threadIdx.x;
  if (i >= DGV * 9 * HWC) return;
  int g = i / (9 * HWC);
  int r = i - g * 9 * HWC;
  int k = r / HWC;
  int p = r - k * HWC;
  int y = p / WV;
  int x = p - y * WV;

  float oy = offs[(g * 18 + k * 2 + 0) * HWC + p];
  float ox = offs[(g * 18 + k * 2 + 1) * HWC + p];
  float m  = msk[(g * 9 + k) * HWC + p];

  float sy = oy + (float)(y + k / 3 - 1);
  float sx = ox + (float)(x + (k % 3) - 1);
  float fy0 = floorf(sy), fx0 = floorf(sx);
  float fy = sy - fy0,    fx = sx - fx0;
  int y0 = (int)fy0, x0 = (int)fx0;
  int y1 = y0 + 1,   x1 = x0 + 1;

  int oky0 = (y0 >= 0) & (y0 < HV);
  int oky1 = (y1 >= 0) & (y1 < HV);
  int okx0 = (x0 >= 0) & (x0 < WV);
  int okx1 = (x1 >= 0) & (x1 < WV);
  int cy0 = min(max(y0, 0), HV - 1), cy1 = min(max(y1, 0), HV - 1);
  int cx0 = min(max(x0, 0), WV - 1), cx1 = min(max(x1, 0), WV - 1);

  float w00 = (1.f - fy) * (1.f - fx) * ((oky0 & okx0) ? m : 0.f);
  float w01 = (1.f - fy) * fx         * ((oky0 & okx1) ? m : 0.f);
  float w10 = fy * (1.f - fx)         * ((oky1 & okx0) ? m : 0.f);
  float w11 = fy * fx                 * ((oky1 & okx1) ? m : 0.f);
  int i00 = cy0 * WV + cx0, i01 = cy0 * WV + cx1;
  int i10 = cy1 * WV + cx0, i11 = cy1 * WV + cx1;

  for (int c = 0; c < 8; ++c) {
    const _Float16* bp = xin + (size_t)(g * 8 + c) * HWC;
    float v = w00 * (float)bp[i00] + w01 * (float)bp[i01] +
              w10 * (float)bp[i10] + w11 * (float)bp[i11];
    cols[(size_t)((g * 8 + c) * 9 + k) * HWC + p] = (_Float16)v;
  }
}

// ---------------------------------------------------------------------------
// Implicit-GEMM conv via WMMA f32_16x16x32_f16.
//   out[Cout][HW] = W[Cout][K] x B[K][HW]
// Block = 256 threads (8 waves). Block tile = 64 rows x 64 pixels, K-step 64.
// Wave (mq = w&3, nq = w>>2): rows [mq*16,mq*16+16), pixels [nq*32, nq*32+32)
//   2 N-subtiles (2 accumulators) x 2 K-subchunks -> 4 v_wmma per barrier pair,
//   64x64x64 MACs per block K-step with one 8KB A-stage.
// LDS layouts give contiguous 16B fragment loads in contiguous k order (A and B alike):
//   A[64][64] row-major: halves 0-15 <- K=16*hi..
//   B[64][64] stored [N][K]: halves 0-15 <- K=16*hi..
// A-tile staged with gfx1250 async global->LDS (b128) when available.
// ---------------------------------------------------------------------------
__global__ __launch_bounds__(256)
void k_conv_wmma(const _Float16* __restrict__ Bsrc,   // activations [Cin][HW] or cols [K][HW]
                 const _Float16* __restrict__ Wf,     // weights f16 [CoutPad][K]
                 const float* __restrict__ bias,      // [Cout] (f32)
                 _Float16* __restrict__ out16,        // f16 output (may be null)
                 float* __restrict__ out32,           // f32 output (fusion path)
                 const float* __restrict__ resid,     // f32 residual for out32 path
                 const _Float16* __restrict__ addsrc, // f16 residual-add for out16 path
                 int Cout, int K, int relu, int pregathered) {
  __shared__ __align__(16) _Float16 As[64 * 64];   // 8 KB
  __shared__ __align__(16) _Float16 Bs[64 * 64];   // 8 KB

  const int tid  = threadIdx.x;
  const int lane = tid & 31;
  const int wv   = tid >> 5;        // wave 0..7
  const int mq   = wv & 3;          // row group: rows [mq*16, mq*16+16)
  const int nq   = wv >> 2;         // pixel half: [nq*32, nq*32+32)
  const int hi   = lane >> 4;
  const int ln   = lane & 15;
  const int n0   = blockIdx.x * 64; // pixel base
  const int mB   = blockIdx.y * 64; // Cout base (padded weights)

  v8f acc0 = {0.f, 0.f, 0.f, 0.f, 0.f, 0.f, 0.f, 0.f};
  v8f acc1 = {0.f, 0.f, 0.f, 0.f, 0.f, 0.f, 0.f, 0.f};

  const int ar = tid >> 2;          // 0..63 weight row within tile
  const int ap = tid & 3;           // quarter-row: halves [ap*16, ap*16+16)
  const int bn = tid & 63;          // pixel within tile
  const int bk = (tid >> 6) << 4;   // K-half chunk base [bk, bk+16)
  const int bp = n0 + bn;
  const int by = bp / WV;
  const int bx = bp - by * WV;

  for (int k0 = 0; k0 < K; k0 += 64) {
    // ---- stage A (weights): 64 rows x 64 halves; each thread 32 bytes ----
    {
      const _Float16* asrc = Wf + (size_t)(mB + ar) * K + k0 + ap * 16;
      _Float16* adst = &As[ar * 64 + ap * 16];
#ifdef ASYNC_LDS
      __builtin_amdgcn_global_load_async_to_lds_b128((gv4i*)asrc, (lv4i*)adst, 0, 0);
      __builtin_amdgcn_global_load_async_to_lds_b128((gv4i*)(asrc + 8),
                                                     (lv4i*)(adst + 8), 0, 0);
#else
      if (k0 + 64 < K) __builtin_prefetch(asrc + 64, 0, 3);
      *(uint4*)adst       = *(const uint4*)asrc;
      *(uint4*)(adst + 8) = *(const uint4*)(asrc + 8);
#endif
    }
    // ---- stage B: 64 pixels x 64 K; each thread 16 halves -> 2 b128 stores
    {
      union { uint4 u[2]; _Float16 h[16]; } pk;
      if (pregathered) {
        #pragma unroll
        for (int j = 0; j < 16; ++j)
          pk.h[j] = Bsrc[(size_t)(k0 + bk + j) * HWC + bp];
      } else {
        #pragma unroll
        for (int j = 0; j < 16; ++j) {
          int k  = k0 + bk + j;
          int c  = k / 9;
          int t2 = k - c * 9;
          int sy = by + t2 / 3 - 1;
          int sx = bx + (t2 % 3) - 1;
          _Float16 v = (_Float16)0.f;
          if (sy >= 0 && sy < HV && sx >= 0 && sx < WV)
            v = Bsrc[(size_t)c * HWC + sy * WV + sx];
          pk.h[j] = v;
        }
      }
      *(uint4*)&Bs[bn * 64 + bk]     = pk.u[0];
      *(uint4*)&Bs[bn * 64 + bk + 8] = pk.u[1];
    }
#ifdef ASYNC_LDS
    __builtin_amdgcn_s_wait_asynccnt(0);
#endif
    __syncthreads();

    const int mr = (mq << 4) + ln;          // A row for this lane
    const int nb = (nq << 5) + ln;          // first B row (pixel) for this lane
    #pragma unroll
    for (int ks = 0; ks < 2; ++ks) {
      union { uint4 u[2]; v16h v; } af, bf0, bf1;
      const int ka = ks * 32;
      af.u[0]  = *(const uint4*)&As[mr * 64 + ka + hi * 16];
      af.u[1]  = *(const uint4*)&As[mr * 64 + ka + hi * 16 + 8];
      bf0.u[0] = *(const uint4*)&Bs[nb * 64 + ka + hi * 16];
      bf0.u[1] = *(const uint4*)&Bs[nb * 64 + ka + hi * 16 + 8];
      bf1.u[0] = *(const uint4*)&Bs[(nb + 16) * 64 + ka + hi * 16];
      bf1.u[1] = *(const uint4*)&Bs[(nb + 16) * 64 + ka + hi * 16 + 8];
      acc0 = __builtin_amdgcn_wmma_f32_16x16x32_f16(false, af.v, false, bf0.v,
                                                    (short)0, acc0, false, false);
      acc1 = __builtin_amdgcn_wmma_f32_16x16x32_f16(false, af.v, false, bf1.v,
                                                    (short)0, acc1, false, false);
    }
    __syncthreads();
  }

  // ---- epilogue: bias (+lrelu) (+f16 add) -> f16, or (+f32 residual) -> f32
  #pragma unroll
  for (int ns = 0; ns < 2; ++ns) {
    const v8f& a = ns ? acc1 : acc0;
    int p = n0 + (nq << 5) + ns * 16 + ln;
    #pragma unroll
    for (int r = 0; r < 8; ++r) {
      int co = mB + (mq << 4) + (hi << 3) + r;
      if (co < Cout) {
        float v = a[r] + bias[co];
        if (relu) v = (v >= 0.f) ? v : 0.1f * v;
        size_t o = (size_t)co * HWC + p;
        if (out32) {
          out32[o] = v + (resid ? resid[o] : 0.f);
        } else {
          if (addsrc) v += (float)addsrc[o];
          out16[o] = (_Float16)v;
        }
      }
    }
  }
}

// ---------------------------------------------------------------------------
// Host orchestration
// ---------------------------------------------------------------------------
extern "C" void kernel_launch(void* const* d_in, const int* in_sizes, int n_in,
                              void* d_out, int out_size, void* d_ws, size_t ws_size,
                              hipStream_t stream) {
  (void)in_sizes; (void)n_in; (void)out_size; (void)ws_size;

  const float* x = (const float*)d_in[0];
  struct Dir {
    const float *dw, *db, *ow1, *ob1, *ow2, *ob2, *ow3, *ob3, *ow4, *ob4,
                *bw1, *bb1, *bw2, *bb2;
  } P[2];
  for (int d = 0; d < 2; ++d) {
    int b = 1 + d * 14;
    P[d].dw  = (const float*)d_in[b + 0];  P[d].db  = (const float*)d_in[b + 1];
    P[d].ow1 = (const float*)d_in[b + 2];  P[d].ob1 = (const float*)d_in[b + 3];
    P[d].ow2 = (const float*)d_in[b + 4];  P[d].ob2 = (const float*)d_in[b + 5];
    P[d].ow3 = (const float*)d_in[b + 6];  P[d].ob3 = (const float*)d_in[b + 7];
    P[d].ow4 = (const float*)d_in[b + 8];  P[d].ob4 = (const float*)d_in[b + 9];
    P[d].bw1 = (const float*)d_in[b + 10]; P[d].bb1 = (const float*)d_in[b + 11];
    P[d].bw2 = (const float*)d_in[b + 12]; P[d].bb2 = (const float*)d_in[b + 13];
  }
  const float* fus_w = (const float*)d_in[29];
  const float* fus_b = (const float*)d_in[30];
  float* dout = (float*)d_out;

  // bump allocator over workspace
  char* ws = (char*)d_ws;
  size_t off = 0;
  auto alloc = [&](size_t bytes) -> char* {
    char* p = ws + off;
    off = (off + bytes + 255) & ~(size_t)255;
    return p;
  };

  _Float16* xs    = (_Float16*)alloc(sizeof(_Float16) * (size_t)TV * CHWn);
  _Float16* featB = (_Float16*)alloc(sizeof(_Float16) * (size_t)TV * CHWn);
  _Float16* featF = (_Float16*)alloc(sizeof(_Float16) * (size_t)TV * CHWn);
  _Float16* prop  = (_Float16*)alloc(sizeof(_Float16) * CHWn);
  _Float16* zbuf  = (_Float16*)alloc(sizeof(_Float16) * CHWn);
  _Float16* cond  = (_Float16*)alloc(sizeof(_Float16) * 3 * (size_t)CHWn);
  _Float16* hA    = (_Float16*)alloc(sizeof(_Float16) * CHWn);
  _Float16* hB    = (_Float16*)alloc(sizeof(_Float16) * CHWn);
  _Float16* o4    = (_Float16*)alloc(sizeof(_Float16) * 432 * (size_t)HWC);
  float*    offs  = (float*)alloc(sizeof(float) * 288 * (size_t)HWC);
  float*    msk   = (float*)alloc(sizeof(float) * 144 * (size_t)HWC);
  _Float16* dcnin = (_Float16*)alloc(sizeof(_Float16) * 2 * (size_t)CHWn);
  _Float16* cols  = (_Float16*)alloc(sizeof(_Float16) * 1152 * (size_t)HWC);

  struct DirW { _Float16 *dw, *ow1, *ow2, *ow3, *ow4, *bw1, *bw2; } WD[2];
  for (int d = 0; d < 2; ++d) {
    WD[d].dw  = (_Float16*)alloc(sizeof(_Float16) * 64 * 1152);
    WD[d].ow1 = (_Float16*)alloc(sizeof(_Float16) * 64 * 1728);
    WD[d].ow2 = (_Float16*)alloc(sizeof(_Float16) * 64 * 576);
    WD[d].ow3 = (_Float16*)alloc(sizeof(_Float16) * 64 * 576);
    WD[d].ow4 = (_Float16*)alloc(sizeof(_Float16) * 448 * 576);   // 432 padded -> 448
    WD[d].bw1 = (_Float16*)alloc(sizeof(_Float16) * 64 * (d ? 1728 : 1152));
    WD[d].bw2 = (_Float16*)alloc(sizeof(_Float16) * 64 * 576);
  }
  _Float16* fusf = (_Float16*)alloc(sizeof(_Float16) * 64 * 128);

  auto cvtw = [&](const float* s, _Float16* dst, int nvalid, int ntot) {
    k_cvtw<<<dim3((ntot + 255) / 256), dim3(256), 0, stream>>>(s, dst, nvalid, ntot);
  };
  for (int d = 0; d < 2; ++d) {
    cvtw(P[d].dw,  WD[d].dw,  64 * 1152, 64 * 1152);
    cvtw(P[d].ow1, WD[d].ow1, 64 * 1728, 64 * 1728);
    cvtw(P[d].ow2, WD[d].ow2, 64 * 576,  64 * 576);
    cvtw(P[d].ow3, WD[d].ow3, 64 * 576,  64 * 576);
    cvtw(P[d].ow4, WD[d].ow4, 432 * 576, 448 * 576);
    cvtw(P[d].bw1, WD[d].bw1, 64 * (d ? 1728 : 1152), 64 * (d ? 1728 : 1152));
    cvtw(P[d].bw2, WD[d].bw2, 64 * 576,  64 * 576);
  }
  cvtw(fus_w, fusf, 64 * 128, 64 * 128);
  {
    int n = TV * CHWn;
    k_cvt16<<<dim3((n + 255) / 256), dim3(256), 0, stream>>>(x, xs, n);
  }

  auto conv = [&](const _Float16* src, const _Float16* wt, const float* bb,
                  _Float16* o16, float* o32, const float* resid, const _Float16* add,
                  int Cout, int K, int relu, int pre) {
    int CoutPad = (Cout + 63) & ~63;
    dim3 g(HWC / 64, CoutPad / 64), b(256);
    k_conv_wmma<<<g, b, 0, stream>>>(src, wt, bb, o16, o32, resid, add, Cout, K, relu, pre);
  };
  auto copy = [&](_Float16* dst, const _Float16* src, int nh) {
    k_copy8<<<dim3((nh / 8 + 255) / 256), dim3(256), 0, stream>>>(dst, src, nh / 8);
  };
  auto fill = [&](_Float16* dst, int nh) {
    k_fill8<<<dim3((nh / 8 + 255) / 256), dim3(256), 0, stream>>>(dst, nh / 8);
  };

  fill(zbuf, CHWn);

  for (int d = 0; d < 2; ++d) {               // d=0: backward, d=1: forward
    _Float16* feats = d ? featF : featB;
    fill(prop, CHWn);
    for (int i = 0; i < TV; ++i) {
      int idx = d ? i : (TV - 1 - i);
      const _Float16* cur = xs + (size_t)idx * CHWn;

      if (i > 0) {
        const _Float16* fn2 = zbuf;
        if (i > 1) {
          int i2 = d ? (i - 2) : (TV - 1 - (i - 2));
          fn2 = feats + (size_t)i2 * CHWn;
        }
        // cond = concat(prop, cur, feat_n2) ; offset/mask conv stack
        copy(cond, prop, CHWn);
        copy(cond + CHWn, cur, CHWn);
        copy(cond + 2 * CHWn, fn2, CHWn);
        conv(cond, WD[d].ow1, P[d].ob1, hA, nullptr, nullptr, nullptr, 64, 1728, 1, 0);
        conv(hA,   WD[d].ow2, P[d].ob2, hB, nullptr, nullptr, nullptr, 64, 576, 1, 0);
        conv(hB,   WD[d].ow3, P[d].ob3, hA, nullptr, nullptr, nullptr, 64, 576, 1, 0);
        conv(hA,   WD[d].ow4, P[d].ob4, o4, nullptr, nullptr, nullptr, 432, 576, 0, 0);
        k_offmask<<<dim3((432 * HWC + 255) / 256), dim3(256), 0, stream>>>(o4, offs, msk);
        // deformable conv input = concat(prop, feat_n2); gather columns; GEMM
        copy(dcnin, prop, CHWn);
        copy(dcnin + CHWn, fn2, CHWn);
        k_cols<<<dim3((DGV * 9 * HWC + 255) / 256), dim3(256), 0, stream>>>(dcnin, offs, msk, cols);
        conv(cols, WD[d].dw, P[d].db, prop, nullptr, nullptr, nullptr, 64, 1152, 0, 1);
      }

      int Kb;
      if (d) {  // forward: feat = concat(cur, feats_backward[idx], prop)
        copy(cond, cur, CHWn);
        copy(cond + CHWn, featB + (size_t)idx * CHWn, CHWn);
        copy(cond + 2 * CHWn, prop, CHWn);
        Kb = 1728;
      } else {  // backward: feat = concat(cur, prop)
        copy(cond, cur, CHWn);
        copy(cond + CHWn, prop, CHWn);
        Kb = 1152;
      }
      conv(cond, WD[d].bw1, P[d].bb1, hA, nullptr, nullptr, nullptr, 64, Kb, 1, 0);
      // prop = prop + conv(lrelu(...)) : fused residual-add epilogue
      conv(hA, WD[d].bw2, P[d].bb2, prop, nullptr, nullptr, prop, 64, 576, 0, 0);
      copy(feats + (size_t)idx * CHWn, prop, CHWn);
    }
  }

  // fusion: 1x1 conv over concat(featB[t], featF[t]) + bias + x[t] -> f32 out
  for (int t = 0; t < TV; ++t) {
    copy(cond, featB + (size_t)t * CHWn, CHWn);
    copy(cond + CHWn, featF + (size_t)t * CHWn, CHWn);
    conv(cond, fusf, fus_b, nullptr, dout + (size_t)t * CHWn, x + (size_t)t * CHWn,
         nullptr, 64, 128, 0, 1);
  }
}
